// SpatialAttention_89988154786232
// MI455X (gfx1250) — matrix operand order repaired and checked by reading, not compiled, not run on hardware
//
#include <hip/hip_runtime.h>

#define BDIM 2
#define CCH  256
#define HH   48
#define WWD  48
#define HWN  2304
#define NH   8
#define HD   32
#define SROW 2312   // 2304 + 8 pad floats per score row

typedef __attribute__((ext_vector_type(16))) __bf16 v16bf;
typedef __attribute__((ext_vector_type(8)))  __bf16 v8bf;
typedef __attribute__((ext_vector_type(8)))  float  v8f;

__device__ __forceinline__ __bf16 f2bf(float f) {
  unsigned int u = __builtin_bit_cast(unsigned int, f);
  u += 0x7FFFu + ((u >> 16) & 1u);          // round-to-nearest-even
  unsigned short s = (unsigned short)(u >> 16);
  return __builtin_bit_cast(__bf16, s);
}

__device__ __forceinline__ float bf2f(__bf16 v) {
  unsigned int u = ((unsigned int)__builtin_bit_cast(unsigned short, v)) << 16;
  return __builtin_bit_cast(float, u);
}

// A operand 16x32 bf16: lane<16 holds row M=lane, K {0..7,16..23}; lane>=16 row M=lane-16, K {8..15,24..31}
// rowptr must already point at A[row = lane&15][k-chunk base], K contiguous.
__device__ __forceinline__ v16bf load_a16x32(const __bf16* rowptr, int lane) {
  const bool hi = lane >= 16;
  const v8bf x = *(const v8bf*)(rowptr + (hi ? 8 : 0));
  const v8bf y = *(const v8bf*)(rowptr + (hi ? 24 : 16));
  v16bf r;
#pragma unroll
  for (int i = 0; i < 8; ++i) { r[i] = x[i]; r[8 + i] = y[i]; }
  return r;
}

// B operand 32x16 bf16: lane<16: N=lane, elements = K {0..7,16..23}; lane>=16: N=lane-16, elements = K {8..15,24..31}
// rowptr must point at Bsrc[N = lane&15][k-chunk base] where source is N-major with K contiguous.
__device__ __forceinline__ v16bf load_b32x16(const __bf16* rowptr, int lane) {
  const __bf16* p = rowptr + ((lane >= 16) ? 8 : 0);
  const v8bf x = *(const v8bf*)(p);
  const v8bf y = *(const v8bf*)(p + 16);
  v16bf r;
#pragma unroll
  for (int i = 0; i < 8; ++i) { r[i] = x[i]; r[8 + i] = y[i]; }
  return r;
}

__device__ __forceinline__ v8f wmma_bf16(v16bf a, v16bf b, v8f c) {
  return __builtin_amdgcn_wmma_f32_16x16x32_bf16(false, a, false, b, (short)0, c, false, false);
}

// ---------------- elementwise / conversion kernels ----------------

__global__ void k_f32_to_bf16(const float* __restrict__ in, __bf16* __restrict__ out, int n) {
  int i = blockIdx.x * 256 + threadIdx.x;
  if (i < n) out[i] = f2bf(in[i]);
}

// in: [B][C][HW] fp32  ->  out: [B][HW][C] bf16
__global__ void k_transpose_cvt(const float* __restrict__ in, __bf16* __restrict__ out) {
  int i = blockIdx.x * 256 + threadIdx.x;
  if (i >= BDIM * CCH * HWN) return;
  int p = i % HWN;
  int c = (i / HWN) % CCH;
  int b = i / (CCH * HWN);
  out[((size_t)b * HWN + p) * CCH + c] = f2bf(in[i]);
}

__global__ void k_zero_f32(float* __restrict__ p, int n) {
  int i = blockIdx.x * 256 + threadIdx.x;
  if (i < n) p[i] = 0.0f;
}

// ---------------- occlusion predictor: im2col + WMMA implicit GEMM ----------------

// depth [B][C][HW] fp32 -> col [B][HW][C*9] bf16, kk = c*9 + (ky*3+kx); zero padding at borders
__global__ void k_im2col1(const float* __restrict__ in, __bf16* __restrict__ col) {
  int i = blockIdx.x * 256 + threadIdx.x;   // p fastest -> coalesced reads
  if (i >= BDIM * CCH * HWN) return;
  int p = i % HWN;
  int c = (i / HWN) % CCH;
  int b = i / (CCH * HWN);
  int y = p / WWD, x = p % WWD;
  const float* ip = in + ((size_t)b * CCH + c) * HWN;
  __bf16* op = col + ((size_t)b * HWN + p) * (CCH * 9) + c * 9;
  int j = 0;
#pragma unroll
  for (int dy = -1; dy <= 1; ++dy) {
#pragma unroll
    for (int dx = -1; dx <= 1; ++dx, ++j) {
      int yy = y + dy, xx = x + dx;
      float val = (yy >= 0 && yy < HH && xx >= 0 && xx < WWD) ? ip[yy * WWD + xx] : 0.0f;
      op[j] = f2bf(val);
    }
  }
}

// h1p [B][HW][128] bf16 (p-major) -> col [B][HW][128*9] bf16, kk = c*9 + j
__global__ void k_im2col2(const __bf16* __restrict__ h1p, __bf16* __restrict__ col) {
  int i = blockIdx.x * 256 + threadIdx.x;
  if (i >= BDIM * 128 * HWN) return;
  int p = i % HWN;
  int c = (i / HWN) % 128;
  int b = i / (128 * HWN);
  int y = p / WWD, x = p % WWD;
  const __bf16 zero = __builtin_bit_cast(__bf16, (unsigned short)0);
  const __bf16* hb = h1p + (size_t)b * HWN * 128 + c;
  __bf16* op = col + ((size_t)b * HWN + p) * (128 * 9) + c * 9;
  int j = 0;
#pragma unroll
  for (int dy = -1; dy <= 1; ++dy) {
#pragma unroll
    for (int dx = -1; dx <= 1; ++dx, ++j) {
      int yy = y + dy, xx = x + dx;
      bool inb = (yy >= 0 && yy < HH && xx >= 0 && xx < WWD);
      op[j] = inb ? hb[(size_t)(yy * WWD + xx) * 128] : zero;
    }
  }
}

// Y[b][p][o] = relu(bn(conv)) as bf16; grid (HWN/16/8, Cout/16, BDIM), 8 waves, 1 tile each
__global__ void k_convgemm_bn_relu(const __bf16* __restrict__ Wb, const __bf16* __restrict__ Colb,
                                   const float* __restrict__ bc, const float* __restrict__ g,
                                   const float* __restrict__ be, const float* __restrict__ m,
                                   const float* __restrict__ v, __bf16* __restrict__ Y,
                                   int K, int Cout) {
  int tid = threadIdx.x, wave = tid >> 5, lane = tid & 31;
  int n0 = (blockIdx.x * 8 + wave) * 16;
  int m0 = blockIdx.y * 16;
  int b  = blockIdx.z;
  const __bf16* wrow = Wb + (size_t)(m0 + (lane & 15)) * K;
  const __bf16* xrow = Colb + ((size_t)b * HWN + n0 + (lane & 15)) * K;
  v8f acc = {};
  for (int kc = 0; kc < K / 32; ++kc) {
    __builtin_prefetch((const void*)(xrow + kc * 32 + 128), 0, 1);
    v16bf a  = load_a16x32(wrow + kc * 32, lane);
    v16bf bb = load_b32x16(xrow + kc * 32, lane);
    acc = wmma_bf16(a, bb, acc);
  }
  int p = n0 + (lane & 15);
  int rbase = (lane < 16) ? 0 : 8;
#pragma unroll
  for (int r = 0; r < 8; ++r) {
    int o = m0 + rbase + r;
    float inv = g[o] * rsqrtf(v[o] + 1e-5f);
    float val = (acc[r] + bc[o]) * inv + (be[o] - m[o] * inv);
    Y[((size_t)b * HWN + p) * Cout + o] = f2bf(fmaxf(val, 0.0f));
  }
}

// vis from h2p [B][HW][64] bf16 (contiguous dot per pixel)
__global__ void k_proj1x1_sigmoid(const __bf16* __restrict__ h2p, const float* __restrict__ w3,
                                  const float* __restrict__ b3, float* __restrict__ vis) {
  int i = blockIdx.x * 256 + threadIdx.x;
  if (i >= BDIM * HWN) return;
  float s = b3[0];
  const __bf16* hb = h2p + (size_t)i * 64;
#pragma unroll 8
  for (int c = 0; c < 64; ++c) s += bf2f(hb[c]) * w3[c];
  vis[i] = 1.0f / (1.0f + __expf(-s));
}

// ---------------- WMMA GEMM: Y[o,p] = sum_c Wb[o,c] * Xt[b,p,c] (+bias, +residual) ----------------
// mode 0: bf16 out, p-major (Q,K); mode 1: bf16 out, o-major (V); mode 2: f32 out o-major + resid
__global__ void k_gemm_wmma(const __bf16* __restrict__ Wb, const __bf16* __restrict__ Xt,
                            const float* __restrict__ bias, const float* __restrict__ resid,
                            void* __restrict__ Yout, int mode) {
  int tid = threadIdx.x, wave = tid >> 5, lane = tid & 31;
  int n0 = (blockIdx.x * 8 + wave) * 16;
  int m0 = blockIdx.y * 16;
  int b  = blockIdx.z;
  const __bf16* wrow = Wb + (size_t)(m0 + (lane & 15)) * CCH;
  const __bf16* xrow = Xt + ((size_t)b * HWN + n0 + (lane & 15)) * CCH;
  v8f acc = {};
#pragma unroll
  for (int kc = 0; kc < CCH / 32; ++kc) {
    v16bf a  = load_a16x32(wrow + kc * 32, lane);
    v16bf bb = load_b32x16(xrow + kc * 32, lane);
    acc = wmma_bf16(a, bb, acc);
  }
  int p = n0 + (lane & 15);
  int rbase = (lane < 16) ? 0 : 8;
  if (mode == 0) {
    __bf16* Y = (__bf16*)Yout;
#pragma unroll
    for (int r = 0; r < 8; ++r) {
      int o = m0 + rbase + r;
      Y[((size_t)b * HWN + p) * CCH + o] = f2bf(acc[r] + bias[o]);
    }
  } else if (mode == 1) {
    __bf16* Y = (__bf16*)Yout;
#pragma unroll
    for (int r = 0; r < 8; ++r) {
      int o = m0 + rbase + r;
      Y[((size_t)b * CCH + o) * HWN + p] = f2bf(acc[r] + bias[o]);
    }
  } else {
    float* Y = (float*)Yout;
#pragma unroll
    for (int r = 0; r < 8; ++r) {
      int o = m0 + rbase + r;
      size_t idx = ((size_t)b * CCH + o) * HWN + p;
      Y[idx] = acc[r] + bias[o] + resid[idx];
    }
  }
}

// ---------------- flash attention: one (b, head, 16-query tile) per block ----------------
__global__ void k_attention(const __bf16* __restrict__ Qb, const __bf16* __restrict__ Kb,
                            const __bf16* __restrict__ Vb, const float* __restrict__ vis,
                            float* __restrict__ accmap, __bf16* __restrict__ At) {
  extern __shared__ float smem[];
  float* otile = smem + 16 * SROW;

  int tid = threadIdx.x, wave = tid >> 5, lane = tid & 31;
  int q0 = blockIdx.x * 16;
  int h  = blockIdx.y;
  int b  = blockIdx.z;

  otile[tid] = 0.0f;
  otile[tid + 256] = 0.0f;

  // ---- S = (Q^T K) * scale * vis, row-block resident in LDS ----
  const __bf16* qrow = Qb + ((size_t)b * HWN + q0 + (lane & 15)) * CCH + h * HD;
  v16bf aq = load_a16x32(qrow, lane);
  const float scale = 0.17677669529663687f; // 1/sqrt(32)
  for (int kt = wave; kt < HWN / 16; kt += 8) {
    int kcol = kt * 16 + (lane & 15);
    const __bf16* krow = Kb + ((size_t)b * HWN + kcol) * CCH + h * HD;
    v16bf bk = load_b32x16(krow, lane);
    v8f cc = {};
    cc = wmma_bf16(aq, bk, cc);
    float vv = vis[b * HWN + kcol] * scale;
    int rbase = (lane < 16) ? 0 : 8;
#pragma unroll
    for (int r = 0; r < 8; ++r) smem[(rbase + r) * SROW + kcol] = cc[r] * vv;
  }
  __syncthreads();

  // ---- exact softmax per query row (16 threads per row) ----
  {
    int row = tid >> 4, sub = tid & 15;
    float* srow = smem + row * SROW;
    float mx = -3.4e38f;
    for (int c = sub; c < HWN; c += 16) mx = fmaxf(mx, srow[c]);
#pragma unroll
    for (int msk = 8; msk >= 1; msk >>= 1) mx = fmaxf(mx, __shfl_xor(mx, msk, 16));
    float sum = 0.0f;
    for (int c = sub; c < HWN; c += 16) {
      float e = __expf(srow[c] - mx);
      srow[c] = e;
      sum += e;
    }
#pragma unroll
    for (int msk = 8; msk >= 1; msk >>= 1) sum += __shfl_xor(sum, msk, 16);
    float inv = 1.0f / sum;
    for (int c = sub; c < HWN; c += 16) srow[c] *= inv;
  }
  __syncthreads();

  // ---- column sums for the attention map ----
  for (int c = tid; c < HWN; c += 256) {
    float s = 0.0f;
#pragma unroll
    for (int r = 0; r < 16; ++r) s += smem[r * SROW + c];
    atomicAdd(&accmap[b * HWN + c], s);
  }

  // ---- O = P V^T via WMMA, k in chunks of 32, waves partial-sum into otile ----
  v8f acc0 = {}, acc1 = {};
  const float* pbase = smem + (lane & 15) * SROW;
  const __bf16* v0base = Vb + ((size_t)b * CCH + h * HD + (lane & 15)) * HWN;
  const __bf16* v1base = Vb + ((size_t)b * CCH + h * HD + 16 + (lane & 15)) * HWN;
  int o0 = (lane < 16) ? 0 : 8, o1 = (lane < 16) ? 16 : 24;
  for (int kc = wave; kc < HWN / 32; kc += 8) {
    const float* pr = pbase + kc * 32;
    v16bf ap;
#pragma unroll
    for (int i = 0; i < 8; ++i) { ap[i] = f2bf(pr[o0 + i]); ap[8 + i] = f2bf(pr[o1 + i]); }
    v16bf b0 = load_b32x16(v0base + kc * 32, lane);
    v16bf b1 = load_b32x16(v1base + kc * 32, lane);
    acc0 = wmma_bf16(ap, b0, acc0);
    acc1 = wmma_bf16(ap, b1, acc1);
  }
  {
    int rbase = (lane < 16) ? 0 : 8, ncol = lane & 15;
#pragma unroll
    for (int r = 0; r < 8; ++r) {
      atomicAdd(&otile[(rbase + r) * 32 + ncol], acc0[r]);
      atomicAdd(&otile[(rbase + r) * 32 + 16 + ncol], acc1[r]);
    }
  }
  __syncthreads();

  // ---- write attended tile, p-major bf16 for the output-projection GEMM ----
#pragma unroll
  for (int e = tid; e < 512; e += 256) {
    int r = e >> 5, d = e & 31;
    At[((size_t)b * HWN + q0 + r) * CCH + h * HD + d] = f2bf(otile[e]);
  }
}

__global__ void k_finalize_map(const float* __restrict__ accmap, const float* __restrict__ vis,
                               float* __restrict__ out) {
  int i = blockIdx.x * 256 + threadIdx.x;
  if (i >= BDIM * HWN) return;
  out[i] = accmap[i] * (1.0f / (8.0f * 2304.0f)) * vis[i];
}

// ---------------- host side ----------------

extern "C" void kernel_launch(void* const* d_in, const int* in_sizes, int n_in,
                              void* d_out, int out_size, void* d_ws, size_t ws_size,
                              hipStream_t stream) {
  const float* rgb   = (const float*)d_in[0];
  const float* depth = (const float*)d_in[1];
  const float* wq = (const float*)d_in[2];  const float* bq = (const float*)d_in[3];
  const float* wk = (const float*)d_in[4];  const float* bk = (const float*)d_in[5];
  const float* wv = (const float*)d_in[6];  const float* bv = (const float*)d_in[7];
  const float* wo = (const float*)d_in[8];  const float* bo = (const float*)d_in[9];
  const float* ow1 = (const float*)d_in[10]; const float* ob1 = (const float*)d_in[11];
  const float* g1 = (const float*)d_in[12]; const float* be1 = (const float*)d_in[13];
  const float* m1 = (const float*)d_in[14]; const float* v1  = (const float*)d_in[15];
  const float* ow2 = (const float*)d_in[16]; const float* ob2 = (const float*)d_in[17];
  const float* g2 = (const float*)d_in[18]; const float* be2 = (const float*)d_in[19];
  const float* m2 = (const float*)d_in[20]; const float* v2  = (const float*)d_in[21];
  const float* ow3 = (const float*)d_in[22]; const float* ob3 = (const float*)d_in[23];

  char* ws = (char*)d_ws;
  size_t off = 0;
  // persistent across both phases
  float* vis = (float*)(ws + off); off += (size_t)BDIM * HWN * 4;
  float* acc = (float*)(ws + off); off += (size_t)BDIM * HWN * 4;
  const size_t base = off;  // 36864, 256-aligned

  // ---- phase A scratch (occlusion predictor), aliased with phase B ----
  size_t pa = base;
  __bf16* w1bf = (__bf16*)(ws + pa); pa += (size_t)128 * 2304 * 2;      // 589824
  __bf16* w2bf = (__bf16*)(ws + pa); pa += (size_t)64 * 1152 * 2;       // 147456
  __bf16* h1p  = (__bf16*)(ws + pa); pa += (size_t)BDIM * HWN * 128 * 2;
  __bf16* h2p  = (__bf16*)(ws + pa); pa += (size_t)BDIM * HWN * 64 * 2;
  __bf16* col  = (__bf16*)(ws + pa); pa += (size_t)BDIM * HWN * 2304 * 2; // reused for both im2cols

  // ---- phase B scratch (attention) ----
  const size_t WMAT = (size_t)CCH * CCH * 2;
  const size_t ACTB = (size_t)BDIM * HWN * CCH * 2;
  size_t pb = base;
  __bf16* Wq_bf = (__bf16*)(ws + pb); pb += WMAT;
  __bf16* Wk_bf = (__bf16*)(ws + pb); pb += WMAT;
  __bf16* Wv_bf = (__bf16*)(ws + pb); pb += WMAT;
  __bf16* Wo_bf = (__bf16*)(ws + pb); pb += WMAT;
  __bf16* Xt = (__bf16*)(ws + pb); pb += ACTB;
  __bf16* Qb = (__bf16*)(ws + pb); pb += ACTB;
  __bf16* Kb = (__bf16*)(ws + pb); pb += ACTB;
  __bf16* Vb = (__bf16*)(ws + pb); pb += ACTB;
  __bf16* At = (__bf16*)(ws + pb); pb += ACTB;

  size_t need = (pa > pb) ? pa : pb;
  if (need > ws_size) return;

  // ================= phase A: occlusion predictor (WMMA implicit GEMM) =================
  const int NW1 = 128 * 2304, NW2 = 64 * 1152;
  k_f32_to_bf16<<<(NW1 + 255) / 256, 256, 0, stream>>>(ow1, w1bf, NW1);
  k_f32_to_bf16<<<(NW2 + 255) / 256, 256, 0, stream>>>(ow2, w2bf, NW2);

  k_im2col1<<<(BDIM * CCH * HWN + 255) / 256, 256, 0, stream>>>(depth, col);
  k_convgemm_bn_relu<<<dim3(HWN / 128, 8, BDIM), 256, 0, stream>>>(
      w1bf, col, ob1, g1, be1, m1, v1, h1p, 2304, 128);
  k_im2col2<<<(BDIM * 128 * HWN + 255) / 256, 256, 0, stream>>>(h1p, col);
  k_convgemm_bn_relu<<<dim3(HWN / 128, 4, BDIM), 256, 0, stream>>>(
      w2bf, col, ob2, g2, be2, m2, v2, h2p, 1152, 64);
  k_proj1x1_sigmoid<<<(BDIM * HWN + 255) / 256, 256, 0, stream>>>(h2p, ow3, ob3, vis);

  // ================= phase B: attention =================
  const int NW = CCH * CCH;
  k_f32_to_bf16<<<(NW + 255) / 256, 256, 0, stream>>>(wq, Wq_bf, NW);
  k_f32_to_bf16<<<(NW + 255) / 256, 256, 0, stream>>>(wk, Wk_bf, NW);
  k_f32_to_bf16<<<(NW + 255) / 256, 256, 0, stream>>>(wv, Wv_bf, NW);
  k_f32_to_bf16<<<(NW + 255) / 256, 256, 0, stream>>>(wo, Wo_bf, NW);

  const int NX = BDIM * CCH * HWN;
  k_transpose_cvt<<<(NX + 255) / 256, 256, 0, stream>>>(rgb, Xt);
  k_zero_f32<<<(BDIM * HWN + 255) / 256, 256, 0, stream>>>(acc, BDIM * HWN);

  dim3 ggrid(HWN / 16 / 8, CCH / 16, BDIM);
  k_gemm_wmma<<<ggrid, 256, 0, stream>>>(Wq_bf, Xt, bq, nullptr, Qb, 0);
  k_gemm_wmma<<<ggrid, 256, 0, stream>>>(Wk_bf, Xt, bk, nullptr, Kb, 0);
  k_gemm_wmma<<<ggrid, 256, 0, stream>>>(Wv_bf, Xt, bv, nullptr, Vb, 1);

  const int SMEM = (16 * SROW + 512) * 4; // ~150 KB dynamic LDS (2 WGs / 320KB WGP)
  hipFuncSetAttribute((const void*)k_attention, hipFuncAttributeMaxDynamicSharedMemorySize, SMEM);
  k_attention<<<dim3(HWN / 16, NH, BDIM), 256, SMEM, stream>>>(Qb, Kb, Vb, vis, acc, At);

  k_gemm_wmma<<<ggrid, 256, 0, stream>>>(Wo_bf, At, bo, rgb, d_out, 2);
  k_finalize_map<<<(BDIM * HWN + 255) / 256, 256, 0, stream>>>(
      acc, vis, (float*)d_out + (size_t)BDIM * CCH * HWN);

  (void)in_sizes; (void)n_in; (void)out_size;
}
